// se3AtomCloudNet_3917010173961
// MI455X (gfx1250) — matrix operand order repaired and checked by reading, not compiled, not run on hardware
//
#include <hip/hip_runtime.h>
#include <hip/hip_bf16.h>
#include <math.h>

typedef __attribute__((ext_vector_type(16))) _Float16 v16h;
typedef __attribute__((ext_vector_type(8)))  _Float16 v8h;
typedef __attribute__((ext_vector_type(8)))  float    v8f;
typedef _Float16 h16;

#define BB     32
#define NN     30
#define NPAIR  (BB*NN*NN)     // 28800
#define EMBD   32
#define KPAD   128            // radial hidden 100 -> pad 128
#define N1PAD  112            // radial hidden out 100 -> pad 112
#define LU     72             // 3*24 (l,u) channels
#define LUPAD  80
#define RESCH  216
#define RESPAD 224

__device__ __forceinline__ float softplus5(float x) {
  // softplus(5x) * 1.405/5 using HW v_exp_f32 / v_log_f32 (f16 GEMMs dominate error)
  float y  = 5.0f * x;
  float sp = fmaxf(y, 0.0f) + __logf(1.0f + __expf(-fabsf(y)));
  return sp * 0.281f;
}

// A-frag (16x32 f16): lane half lh holds 16 contiguous halves:
//   K in [16*lh, 16*lh+16)  -> two b128 loads + shuffle.
__device__ __forceinline__ v16h load_afrag(const h16* rowptr, int s, int lh) {
  v8h lo = *(const v8h*)(rowptr + 32*s + 16*lh);
  v8h hi = *(const v8h*)(rowptr + 32*s + 16*lh + 8);
  return __builtin_shufflevector(lo, hi, 0,1,2,3,4,5,6,7,8,9,10,11,12,13,14,15);
}
// B operands are stored fragment-swizzled: [s][col][lh][16 halves], where the
// 16 halves cover K = 32*s + 16*lh + e  -> one contiguous 32B load per lane.
__device__ __forceinline__ size_t bswz(int s, int ncols, int col, int lh) {
  return ((size_t)(s*ncols + col)*2 + lh)*16;
}

// -------- K0: pre-swizzle w_res (216x216 f32) into padded f16 fragment layout
__global__ void k0_prep(const float* __restrict__ w_res, h16* __restrict__ w_res_sw) {
  int i = blockIdx.x*256 + threadIdx.x;      // 7*224*2*16 = 50176 entries
  if (i >= 7*RESPAD*32) return;
  int e = i & 15, lh = (i >> 4) & 1;
  int rest = i >> 5;
  int col = rest % RESPAD, s = rest / RESPAD;
  int kk = 32*s + 16*lh + e;
  w_res_sw[i] = (h16)((kk < RESCH && col < RESCH) ? w_res[kk*RESCH + col] : 0.0f);
}

// -------- K1: pairwise geometry, SH angular terms, radial basis, first MLP layer (K=3)
__global__ void k1_geom(const float* __restrict__ xyz, const float* __restrict__ w_r0,
                        float* __restrict__ Abuf, h16* __restrict__ h0) {
  int p = blockIdx.x;
  int z = p / (NN*NN); int rem = p % (NN*NN); int a = rem / NN; int b = rem % NN;
  const float* pa = xyz + (size_t)(z*NN + a)*3;
  const float* pb = xyz + (size_t)(z*NN + b)*3;
  float dx = pb[0]-pa[0], dy = pb[1]-pa[1], dz = pb[2]-pa[2];
  float d2 = dx*dx + dy*dy + dz*dz;
  float nz = (d2 > 0.0f) ? 1.0f : 0.0f;
  float d  = (d2 > 0.0f) ? sqrtf(d2) : 0.0f;
  float mask = (d < 2.0f) ? 1.0f : 0.0f;
  float inv = 1.0f / fmaxf(d, 1e-12f);
  float ux = dx*inv, uy = dy*inv, uz = dz*inv;

  float bas[3];                               // cosine basis, centers {0,1,2}, step 1
  #pragma unroll
  for (int j = 0; j < 3; ++j) {
    float xk = d - (float)j;
    float cv = __cosf(1.57079632679f * xk);
    bas[j] = (fabsf(xk) < 1.0f) ? cv*cv : 0.0f;
  }
  int t = threadIdx.x;
  if (t < 100) {
    float acc = bas[0]*w_r0[t] + bas[1]*w_r0[100 + t] + bas[2]*w_r0[200 + t];
    h0[(size_t)p*KPAD + t] = (h16)softplus5(acc);
  } else {
    h0[(size_t)p*KPAD + t] = (h16)0.0f;       // K padding exact zero
  }
  if (t < 9) {
    const float s3 = 1.7320508075688772f;
    float Av[9];
    Av[0] = 1.0f;
    Av[1] = uy*nz; Av[2] = uz*nz; Av[3] = ux*nz;
    Av[4] = s3*ux*uy*nz; Av[5] = s3*uy*uz*nz;
    Av[6] = 0.5f*(3.0f*uz*uz - 1.0f)*nz;
    Av[7] = s3*ux*uz*nz; Av[8] = 0.5f*s3*(ux*ux - uy*uy)*nz;
    Abuf[(size_t)p*9 + t] = Av[t]*mask;
  }
}

// -------- K2: h1 = softplus5(h0 @ w_r1), w_r1 staged swizzled in LDS
__global__ void k2_mlp1(const h16* __restrict__ h0, const float* __restrict__ w_r1,
                        h16* __restrict__ h1) {
  __shared__ __align__(32) h16 sB[KPAD*N1PAD];   // 4 ksteps * 112 cols * 2 lh * 16
  int tid = threadIdx.x;
  for (int i = tid; i < 4*N1PAD*32; i += 256) {
    int e = i & 15, lh = (i >> 4) & 1;
    int rest = i >> 5;
    int col = rest % N1PAD, s = rest / N1PAD;
    int kk = 32*s + 16*lh + e;
    sB[i] = (h16)((kk < 100 && col < 100) ? w_r1[kk*100 + col] : 0.0f);
  }
  __syncthreads();
  int wid = tid >> 5, lane = tid & 31;
  int gid = blockIdx.x*8 + wid;               // 1800 * 7 = 12600 tiles
  int mt = gid / 7, nt = gid % 7;
  if (mt >= 1800) return;
  int r = lane & 15, lh = lane >> 4;
  int col = nt*16 + r;
  const h16* arow = h0 + (size_t)(mt*16 + r)*KPAD;
  __builtin_prefetch(arow, 0, 1);             // global_prefetch_b8
  v8f acc = {};
  #pragma unroll
  for (int s = 0; s < 4; ++s) {
    v16h af = load_afrag(arow, s, lh);
    v16h bf = *(const v16h*)(sB + bswz(s, N1PAD, col, lh));
    acc = __builtin_amdgcn_wmma_f32_16x16x32_f16(false, af, false, bf, (short)0, acc, false, false);
  }
  #pragma unroll
  for (int i = 0; i < 8; ++i) {
    int row = mt*16 + i + 8*lh;
    float v = (col < 100) ? softplus5(acc[i]) : 0.0f;   // keep pad cols exactly zero
    h1[(size_t)row*KPAD + col] = (h16)v;
  }
}

// -------- K3: W2f[zb][kk][lu] = sum_v w_r2[kk, lu*32+v] * emb[features[zb]][v]
// GEMM M=7200 (kk*72+lu), K=32 (v), N=960 (zb); output written B-frag swizzled for K4
__global__ void k3_w2f(const float* __restrict__ w_r2, const int* __restrict__ features,
                       const float* __restrict__ emb, h16* __restrict__ W2f) {
  int tid = threadIdx.x; int wid = tid >> 5, lane = tid & 31;
  int gid = blockIdx.x*8 + wid;               // 450 * 60 = 27000 tiles
  if (gid >= 450*60) return;
  int mt = gid / 60, nt = gid % 60;
  int r = lane & 15, lh = lane >> 4;
  int row = mt*16 + r;                        // klu
  int k = row / LU, lu = row % LU;
  int col = nt*16 + r;                        // zb
  int fidx = features[col];
  // A frag: two contiguous 8-float runs of w_r2, convert to f16
  v8f w0 = *(const v8f*)(w_r2 + (size_t)k*2304 + lu*32 + 16*lh);
  v8f w1 = *(const v8f*)(w_r2 + (size_t)k*2304 + lu*32 + 16*lh + 8);
  v16h af, bf;
  #pragma unroll
  for (int e = 0; e < 8; ++e) { af[e] = (h16)w0[e]; af[e+8] = (h16)w1[e]; }
  // B frag: 16 contiguous floats of emb row, convert
  v8f e0 = *(const v8f*)(emb + fidx*EMBD + 16*lh);
  v8f e1 = *(const v8f*)(emb + fidx*EMBD + 16*lh + 8);
  #pragma unroll
  for (int e = 0; e < 8; ++e) { bf[e] = (h16)e0[e]; bf[e+8] = (h16)e1[e]; }
  v8f acc = {};
  acc = __builtin_amdgcn_wmma_f32_16x16x32_f16(false, af, false, bf, (short)0, acc, false, false);
  // scatter into swizzled layout: [zb][s][lu-col(80)][lh][e], kk = 32s+16lh+e
  h16* Bz = W2f + (size_t)col*(4*LUPAD*32);
  #pragma unroll
  for (int i = 0; i < 8; ++i) {
    int rr = mt*16 + i + 8*lh;
    int kk = rr / LU, ll = rr % LU;
    int s2 = kk >> 5, r5 = kk & 31, lh2 = r5 >> 4, e2 = r5 & 15;
    Bz[bswz(s2, LUPAD, ll, lh2) + e2] = (h16)acc[i];
  }
}

// -------- K4: T[(z,a,b)][lu] = sum_k h1[(z,a,b)][k] * W2f[zb][k][lu]
__global__ void k4_T(const h16* __restrict__ h1, const h16* __restrict__ W2f,
                     float* __restrict__ Tbuf) {
  int tid = threadIdx.x; int wid = tid >> 5, lane = tid & 31;
  int gid = blockIdx.x*8 + wid;               // 960 * 2 * 5 = 9600 tiles
  int zb = gid / 10; int r10 = gid % 10; int mt = r10 / 5; int nt = r10 % 5;
  int z = zb / NN, b = zb % NN;
  int r = lane & 15, lh = lane >> 4;
  int arow = mt*16 + r;                       // atom index a
  int col = nt*16 + r;
  const h16* Bp = W2f + (size_t)zb*(4*LUPAD*32);
  __builtin_prefetch(Bp, 0, 1);
  const h16* arp = (arow < NN) ? (h1 + (size_t)(z*900 + arow*30 + b)*KPAD) : nullptr;
  const v16h zfrag = {};
  v8f acc = {};
  #pragma unroll
  for (int s = 0; s < 4; ++s) {
    v16h af = arp ? load_afrag(arp, s, lh) : zfrag;
    v16h bf = *(const v16h*)(Bp + bswz(s, LUPAD, col, lh));
    acc = __builtin_amdgcn_wmma_f32_16x16x32_f16(false, af, false, bf, (short)0, acc, false, false);
  }
  #pragma unroll
  for (int i = 0; i < 8; ++i) {
    int a = mt*16 + i + 8*lh;
    if (a < NN && col < LU)
      Tbuf[(size_t)(z*900 + a*30 + b)*LU + col] = acc[i];
  }
}

// -------- K5: SH-weighted neighbor sum -> feat216; also emit f16 padded copy
__global__ void k5_gather(const float* __restrict__ Tbuf, const float* __restrict__ Abuf,
                          float* __restrict__ feat216, h16* __restrict__ feat16) {
  int za = blockIdx.x; int z = za / NN, a = za % NN;
  int c = threadIdx.x;                        // 224 threads
  if (c >= RESCH) { if (c < RESPAD) feat16[(size_t)za*RESPAD + c] = (h16)0.0f; return; }
  int l, u, m;
  if (c < 24)      { l = 0; u = c;            m = 0; }
  else if (c < 96) { int rr = c - 24; l = 1; u = rr/3; m = rr%3; }
  else             { int rr = c - 96; l = 2; u = rr/5; m = rr%5; }
  int sh = (l == 0) ? 0 : ((l == 1) ? 1 : 4);
  int lu = l*24 + u;
  float acc = 0.0f;
  for (int b = 0; b < NN; ++b) {
    size_t p = (size_t)(z*900 + a*30 + b);
    acc += Tbuf[p*LU + lu] * Abuf[p*9 + sh + m];
  }
  feat216[(size_t)za*RESCH + c] = acc;
  feat16 [(size_t)za*RESPAD + c] = (h16)acc;
}

// -------- K6: residual linear relu(feat216 @ w_res + b_res), w_res pre-swizzled f16
__global__ void k6_res(const h16* __restrict__ feat16, const h16* __restrict__ w_res_sw,
                       const float* __restrict__ b_res, float* __restrict__ tlin) {
  int tid = threadIdx.x; int wid = tid >> 5, lane = tid & 31;
  int gid = blockIdx.x*8 + wid;               // 60 * 14 = 840 tiles
  if (gid >= 60*14) return;
  int mt = gid / 14, nt = gid % 14;
  int r = lane & 15, lh = lane >> 4;
  int col = nt*16 + r;
  const h16* arow = feat16 + (size_t)(mt*16 + r)*RESPAD;
  __builtin_prefetch(arow, 0, 1);
  v8f acc = {};
  for (int s = 0; s < 7; ++s) {
    v16h af = load_afrag(arow, s, lh);
    v16h bf = *(const v16h*)(w_res_sw + bswz(s, RESPAD, col, lh));
    acc = __builtin_amdgcn_wmma_f32_16x16x32_f16(false, af, false, bf, (short)0, acc, false, false);
  }
  if (col < RESCH) {
    #pragma unroll
    for (int i = 0; i < 8; ++i) {
      int row = mt*16 + i + 8*lh;             // < 960 always
      float v = acc[i] + b_res[col];
      tlin[(size_t)row*RESCH + col] = fmaxf(v, 0.0f);
    }
  }
}

// -------- K7: per-(atom,channel) BatchNorm over batch axis, then relu
__global__ void k7_bn1(const float* __restrict__ tlin, const float* __restrict__ g,
                       const float* __restrict__ bb, float* __restrict__ t2) {
  int n = blockIdx.x; int c = threadIdx.x;
  if (c >= RESCH) return;
  float m = 0.0f, v = 0.0f;
  for (int z = 0; z < BB; ++z) {
    float x = tlin[(size_t)(z*NN + n)*RESCH + c];
    m += x; v += x*x;
  }
  m *= (1.0f/BB); v = v*(1.0f/BB) - m*m;
  float inv = rsqrtf(v + 1e-5f);
  float gc = g[c], bc = bb[c];
  for (int z = 0; z < BB; ++z) {
    float x = tlin[(size_t)(z*NN + n)*RESCH + c];
    float y = gc*(x - m)*inv + bc;
    t2[(size_t)(z*NN + n)*RESCH + c] = fmaxf(y, 0.0f);
  }
}

// -------- K8: mean pool over atoms of concat(feat216, t2)
__global__ void k8_pool(const float* __restrict__ feat216, const float* __restrict__ t2,
                        float* __restrict__ pooled) {
  int z = blockIdx.x; int c = threadIdx.x;    // 448 threads
  if (c >= 432) return;
  const float* src = (c < RESCH) ? feat216 : t2;
  int cc = (c < RESCH) ? c : (c - RESCH);
  float acc = 0.0f;
  for (int n = 0; n < NN; ++n) acc += src[(size_t)(z*NN + n)*RESCH + cc];
  pooled[z*432 + c] = acc * (1.0f/NN);
}

// -------- K9: head: linear->leaky->BN(batch)->leaky->linear->sigmoid (single block)
__global__ void k9_head(const float* __restrict__ pooled, const float* __restrict__ w_c,
                        const float* __restrict__ b_c, const float* __restrict__ g2,
                        const float* __restrict__ b2, const float* __restrict__ w_o,
                        const float* __restrict__ b_o, float* __restrict__ out) {
  __shared__ float hbuf[BB*128];
  int c = threadIdx.x;                        // 128 threads
  float col[BB];
  for (int z = 0; z < BB; ++z) {
    float acc = b_c[c];
    for (int j = 0; j < 432; ++j) acc += pooled[z*432 + j]*w_c[j*128 + c];
    col[z] = (acc >= 0.0f) ? acc : 0.01f*acc;
  }
  float m = 0.0f, v = 0.0f;
  for (int z = 0; z < BB; ++z) { m += col[z]; v += col[z]*col[z]; }
  m *= (1.0f/BB); v = v*(1.0f/BB) - m*m;
  float inv = rsqrtf(v + 1e-5f);
  float gc = g2[c], bc = b2[c];
  for (int z = 0; z < BB; ++z) {
    float y = gc*(col[z] - m)*inv + bc;
    hbuf[z*128 + c] = (y >= 0.0f) ? y : 0.01f*y;
  }
  __syncthreads();
  if (c < BB) {
    float acc = b_o[0];
    for (int j = 0; j < 128; ++j) acc += hbuf[c*128 + j]*w_o[j];
    out[c] = 1.0f/(1.0f + __expf(-acc));
  }
}

extern "C" void kernel_launch(void* const* d_in, const int* in_sizes, int n_in,
                              void* d_out, int out_size, void* d_ws, size_t ws_size,
                              hipStream_t stream) {
  (void)in_sizes; (void)n_in; (void)out_size; (void)ws_size;
  const float* xyz      = (const float*)d_in[0];
  const int*   features = (const int*)  d_in[1];
  const float* emb      = (const float*)d_in[2];
  const float* w_r0     = (const float*)d_in[3];
  const float* w_r1     = (const float*)d_in[4];
  const float* w_r2     = (const float*)d_in[5];
  const float* w_res    = (const float*)d_in[6];
  const float* b_res    = (const float*)d_in[7];
  const float* bn1_g    = (const float*)d_in[8];
  const float* bn1_b    = (const float*)d_in[9];
  const float* w_c      = (const float*)d_in[10];
  const float* b_c      = (const float*)d_in[11];
  const float* bn2_g    = (const float*)d_in[12];
  const float* bn2_b    = (const float*)d_in[13];
  const float* w_o      = (const float*)d_in[14];
  const float* b_o      = (const float*)d_in[15];
  float* out = (float*)d_out;

  char* ws = (char*)d_ws;
  size_t off = 0;
  auto alloc = [&](size_t bytes) -> char* {
    char* p = ws + off; off += (bytes + 255) & ~(size_t)255; return p;
  };
  float* Abuf     = (float*)alloc((size_t)NPAIR*9*sizeof(float));
  h16*   h0       = (h16*)  alloc((size_t)NPAIR*KPAD*sizeof(h16));
  h16*   h1       = (h16*)  alloc((size_t)NPAIR*KPAD*sizeof(h16));
  h16*   W2f      = (h16*)  alloc((size_t)960*4*LUPAD*32*sizeof(h16));
  float* Tbuf     = (float*)alloc((size_t)NPAIR*LU*sizeof(float));
  float* feat216  = (float*)alloc((size_t)960*RESCH*sizeof(float));
  h16*   feat16   = (h16*)  alloc((size_t)960*RESPAD*sizeof(h16));
  h16*   w_res_sw = (h16*)  alloc((size_t)7*RESPAD*32*sizeof(h16));
  float* tlin     = (float*)alloc((size_t)960*RESCH*sizeof(float));
  float* t2       = (float*)alloc((size_t)960*RESCH*sizeof(float));
  float* pooled   = (float*)alloc((size_t)BB*432*sizeof(float));

  // zero pad regions that WMMA tiles read but kernels don't fully write
  (void)hipMemsetAsync(h1,  0, (size_t)NPAIR*KPAD*sizeof(h16), stream);
  (void)hipMemsetAsync(W2f, 0, (size_t)960*4*LUPAD*32*sizeof(h16), stream);

  k0_prep  <<<196,  256, 0, stream>>>(w_res, w_res_sw);
  k1_geom  <<<NPAIR, 128, 0, stream>>>(xyz, w_r0, Abuf, h0);
  k2_mlp1  <<<1575, 256, 0, stream>>>(h0, w_r1, h1);
  k3_w2f   <<<3375, 256, 0, stream>>>(w_r2, features, emb, W2f);
  k4_T     <<<1200, 256, 0, stream>>>(h1, W2f, Tbuf);
  k5_gather<<<960,  224, 0, stream>>>(Tbuf, Abuf, feat216, feat16);
  k6_res   <<<105,  256, 0, stream>>>(feat16, w_res_sw, b_res, tlin);
  k7_bn1   <<<NN,   224, 0, stream>>>(tlin, bn1_g, bn1_b, t2);
  k8_pool  <<<BB,   448, 0, stream>>>(feat216, t2, pooled);
  k9_head  <<<1,    128, 0, stream>>>(pooled, w_c, b_c, bn2_g, bn2_b, w_o, b_o, out);
}
